// GCN3_21792664060279
// MI455X (gfx1250) — matrix operand order repaired and checked by reading, not compiled, not run on hardware
//
#include <hip/hip_runtime.h>

typedef __attribute__((ext_vector_type(16))) __bf16 v16bf;
typedef __attribute__((ext_vector_type(8)))  float  v8f;

static __device__ __forceinline__ unsigned short f2bfu(float f) {
  unsigned u = __builtin_bit_cast(unsigned, f);
  u += 0x7fffu + ((u >> 16) & 1u);          // round-to-nearest-even
  return (unsigned short)(u >> 16);
}
static __device__ __forceinline__ __bf16 f2bf(float f) {
  unsigned short h = f2bfu(f);
  return __builtin_bit_cast(__bf16, h);
}

// ---------------------------------------------------------------------------
// Repack f32 weights [Co, Ci*9] into WMMA A-fragment order:
// dst[((mt*Kc + kc)*32 + lane)*16 + e]  (bf16, zero-padded K), so a lane's
// fragment is one contiguous 32-byte load.
// ---------------------------------------------------------------------------
__global__ void repack_w(const float* __restrict__ w, __bf16* __restrict__ dst,
                         int Co, int K) {
  const int Kc = (K + 31) >> 5;
  const long total = (long)(Co >> 4) * Kc * 32 * 16;
  long i = (long)blockIdx.x * blockDim.x + threadIdx.x;
  if (i >= total) return;
  int e    = (int)(i & 15);
  int lane = (int)((i >> 4) & 31);
  long t   = i >> 9;
  int kc = (int)(t % Kc);
  int mt = (int)(t / Kc);
  int g = lane >> 4, l15 = lane & 15;
  int kl = (g << 4) + e;
  int k  = (kc << 5) + kl;
  int co = (mt << 4) + l15;
  float v = (k < K) ? w[(long)co * K + k] : 0.f;
  dst[i] = f2bf(v);
}

// ---------------------------------------------------------------------------
// Implicit-GEMM 3x3 conv, stride 1, pad 1.  in:[N,Ci,H,W] f32,
// wpk: repacked bf16 fragments, out:[N,Co,H,W] f32.
// Block: (Co/16) waves, 64 output pixels (4 N-tiles).  K in chunks of 32.
// LDS B tile is pixel-major so fragments are contiguous 32B -> ds_load_b128.
// ---------------------------------------------------------------------------
__global__ void conv3x3_wmma(const float* __restrict__ in, const __bf16* __restrict__ wpk,
                             float* __restrict__ out, int N, int Ci, int Co, int Hi, int Wi) {
  const int K  = Ci * 9;
  const int Kc = (K + 31) >> 5;
  const int HW = Hi * Wi;
  const long P = (long)N * HW;
  const int tid  = threadIdx.x;
  const int lane = tid & 31;
  const int wv   = tid >> 5;             // M tile index
  const int g    = lane >> 4;            // lane group
  const int l15  = lane & 15;
  const long pbase = (long)blockIdx.x * 64;

  // [pixel][kl] packed as u32 pairs: Bsu[p*16 + kl/2]
  __shared__ __attribute__((aligned(32))) unsigned Bsu[64 * 16];

  // staging: each thread owns one pixel; coords computed once
  const int p    = tid & 63;
  const int seg  = tid >> 6;             // 0..(blockDim/64 - 1)
  const int nseg = (int)(blockDim.x >> 6);
  const int span = 32 / (nseg > 0 ? nseg : 1);
  const long gp  = pbase + p;
  const bool pvalid = gp < P;
  int n = 0, y = 0, x = 0;
  if (pvalid) {
    n = (int)(gp / HW);
    int rem = (int)(gp - (long)n * HW);
    y = rem / Wi; x = rem - y * Wi;
  }
  const float* inb = in + (long)n * Ci * HW;
  const v16bf* wfr = (const v16bf*)wpk;

  v8f acc[4] = {};

  for (int kc = 0; kc < Kc; ++kc) {
    const int kbase = kc << 5;
    // ---- stage im2col B tile (2 bf16 per LDS store) ----
    for (int kl = seg * span; kl < seg * span + span; kl += 2) {
      unsigned pk = 0u;
      if (pvalid) {
        #pragma unroll
        for (int q = 0; q < 2; ++q) {
          int k = kbase + kl + q;
          float v = 0.f;
          if (k < K) {
            int ci = k / 9; int r = k - ci * 9;
            int r3 = r / 3;
            int dy = r3 - 1, dx = (r - r3 * 3) - 1;
            int iy = y + dy, ix = x + dx;
            if (iy >= 0 && iy < Hi && ix >= 0 && ix < Wi)
              v = inb[ci * HW + iy * Wi + ix];
          }
          pk |= (unsigned)f2bfu(v) << (16 * q);
        }
      }
      Bsu[p * 16 + (kl >> 1)] = pk;
    }
    __syncthreads();

    // ---- A fragment: one contiguous 32B global load ----
    v16bf a = wfr[((long)wv * Kc + kc) * 32 + lane];

    // ---- 4 B fragments (contiguous 32B LDS loads) + 4 WMMAs ----
    #pragma unroll
    for (int t = 0; t < 4; ++t) {
      const v16bf b = *(const v16bf*)(Bsu + ((t << 4) + l15) * 16 + (g << 3));
      acc[t] = __builtin_amdgcn_wmma_f32_16x16x32_bf16(
          false, a, false, b, (short)0, acc[t], false, false);
    }
    __syncthreads();
  }

  // ---- store C (VGPR r: M=r for lanes 0-15, M=r+8 for lanes 16-31) ----
  const int co_base = (wv << 4) + (g << 3);
  for (int t = 0; t < 4; ++t) {
    long gpt = pbase + (t << 4) + l15;
    if (gpt < P) {
      int nn = (int)(gpt / HW); int rem = (int)(gpt - (long)nn * HW);
      #pragma unroll
      for (int r = 0; r < 8; ++r)
        out[((long)nn * Co + (co_base + r)) * HW + rem] = acc[t][r];
    }
  }
}

// ---------------------------------------------------------------------------
// Training-mode BN statistics: one block per channel -> fused scale/shift.
// ---------------------------------------------------------------------------
__global__ void bn_stats(const float* __restrict__ conv, float* __restrict__ ss,
                         const float* __restrict__ gamma, const float* __restrict__ beta,
                         int N, int Co, int HW) {
  const int co = blockIdx.x;
  double s = 0.0, s2 = 0.0;
  for (int n = 0; n < N; ++n) {
    const float* base = conv + ((long)n * Co + co) * HW;
    for (int j = threadIdx.x; j < HW; j += blockDim.x) {
      float v = base[j];
      s += v; s2 += (double)v * v;
    }
  }
  __shared__ double sh0[256], sh1[256];
  sh0[threadIdx.x] = s; sh1[threadIdx.x] = s2;
  __syncthreads();
  for (int off = blockDim.x >> 1; off > 0; off >>= 1) {
    if ((int)threadIdx.x < off) { sh0[threadIdx.x] += sh0[threadIdx.x + off];
                                  sh1[threadIdx.x] += sh1[threadIdx.x + off]; }
    __syncthreads();
  }
  if (threadIdx.x == 0) {
    const long cnt = (long)N * HW;
    double mean = sh0[0] / (double)cnt;
    double var  = sh1[0] / (double)cnt - mean * mean;
    float sc = gamma[co] * rsqrtf((float)var + 1e-5f);
    ss[co]      = sc;
    ss[Co + co] = beta[co] - (float)mean * sc;
  }
}

__global__ void bn_relu_pool(const float* __restrict__ conv, const float* __restrict__ ss,
                             float* __restrict__ out, int N, int Co, int Hi, int Wi) {
  const int Ho = Hi >> 1, Wo = Wi >> 1;
  const long total = (long)N * Co * Ho * Wo;
  long i = (long)blockIdx.x * blockDim.x + threadIdx.x;
  if (i >= total) return;
  int xo = (int)(i % Wo); long t = i / Wo;
  int yo = (int)(t % Ho); t /= Ho;
  int co = (int)(t % Co); int n = (int)(t / Co);
  float sc = ss[co], sh = ss[Co + co];
  const float* base = conv + ((long)n * Co + co) * Hi * Wi;
  float m = 0.0f;                         // ReLU outputs are >= 0
  for (int dy = 0; dy < 2; ++dy)
    for (int dx = 0; dx < 2; ++dx) {
      float v = base[(yo * 2 + dy) * Wi + (xo * 2 + dx)] * sc + sh;
      v = v > 0.f ? v : 0.f;
      m = v > m ? v : m;
    }
  out[i] = m;
}

__global__ void zero_f32(float* p, long n) {
  long i = (long)blockIdx.x * blockDim.x + threadIdx.x;
  if (i < n) p[i] = 0.f;
}
__global__ void copy_f32(const float* s, float* d, long n) {
  long i = (long)blockIdx.x * blockDim.x + threadIdx.x;
  if (i < n) d[i] = s[i];
}

__global__ void node_norm(const float* __restrict__ x, float* __restrict__ nrm, int Nn) {
  int i = blockIdx.x * blockDim.x + threadIdx.x;
  if (i >= Nn) return;
  float s = 0.f;
  for (int f = 0; f < 64; ++f) { float v = x[i * 64 + f]; s += v * v; }
  float r = sqrtf(s);
  nrm[i] = r > 1e-6f ? r : 1e-6f;
}

__global__ void edge_cos(const float* __restrict__ x, const float* __restrict__ nrm,
                         const int* __restrict__ ei, int E,
                         float* cos_sum, float* cos_cnt) {
  int e = blockIdx.x * blockDim.x + threadIdx.x;
  if (e >= E) return;
  int src = ei[e], dst = ei[E + e];
  float d = 0.f;
  for (int f = 0; f < 64; ++f) d += x[src * 64 + f] * x[dst * 64 + f];
  float c = d / (nrm[src] * nrm[dst]);
  atomicAdd(&cos_sum[dst], c);
  atomicAdd(&cos_cnt[dst], 1.0f);
}

__global__ void support_graph_accum(const float* __restrict__ x,
                                    const float* __restrict__ cos_sum,
                                    const float* __restrict__ cos_cnt,
                                    const int* __restrict__ batch, int Nn,
                                    float* cat_sum, float* cnt, float* scos_sum) {
  int i = blockIdx.x * blockDim.x + threadIdx.x;
  if (i >= Nn) return;
  int b = batch[i];
  for (int f = 0; f < 64; ++f) atomicAdd(&cat_sum[b * 64 + f], x[i * 64 + f]);
  atomicAdd(&cnt[b], 1.0f);
  float cc = cos_cnt[i]; cc = cc > 1.f ? cc : 1.f;
  atomicAdd(&scos_sum[b], cos_sum[i] / cc);
}

__global__ void query_graph_accum(const float* __restrict__ x,
                                  const int* __restrict__ batch, int Nn,
                                  float* cat_sum, float* cnt) {
  int i = blockIdx.x * blockDim.x + threadIdx.x;
  if (i >= Nn) return;
  int b = batch[i];
  for (int f = 0; f < 64; ++f) atomicAdd(&cat_sum[b * 64 + f], x[i * 64 + f]);
  atomicAdd(&cnt[b], 1.0f);
}

__global__ void graph_div(float* cat, const float* cnt, float* scos, int G) {
  int i = blockIdx.x * blockDim.x + threadIdx.x;
  if (i >= G * 64) return;
  int b = i >> 6;
  float c = cnt[b]; c = c > 1.f ? c : 1.f;
  cat[i] = cat[i] / c;
  if (scos != nullptr && (i & 63) == 0) scos[b] = scos[b] / c;
}

// ---------------------------------------------------------------------------
// Tail math in one block.  Outputs: loss, acc, cos_loss, atten_loss, rest_loss.
// ---------------------------------------------------------------------------
__global__ void finalize(const float* __restrict__ s_cat, const float* __restrict__ p_cat,
                         const float* __restrict__ scos_g,
                         const int* __restrict__ support_y, const int* __restrict__ query_y,
                         float* __restrict__ out5) {
  __shared__ float pd1[24 * 120];
  __shared__ float outm[24 * 24];
  __shared__ float ce[24];
  __shared__ int   corr[24];
  const int t = threadIdx.x;

  for (int e = t; e < 24 * 120; e += blockDim.x) {
    int i = e / 120, j = e - i * 120;
    float d2 = 0.f;
    for (int f = 0; f < 64; ++f) {
      float d = s_cat[j * 64 + f] - p_cat[i * 64 + f];
      d2 += d * d;
    }
    pd1[e] = __expf(-d2);
  }
  __syncthreads();

  for (int e = t; e < 24 * 24; e += blockDim.x) {
    int i = e / 24, c = e - i * 24;
    float s = 0.f, n = 0.f;
    for (int j = 0; j < 120; ++j)
      if (support_y[j] == c) { s += pd1[i * 120 + j]; n += 1.f; }
    float nn = n > 1.f ? n : 1.f;
    outm[e] = s / nn;
  }
  __syncthreads();

  if (t < 24) {
    float mx = -1e30f; int am = 0;
    for (int c = 0; c < 24; ++c) {
      float v = outm[t * 24 + c];
      if (v > mx) { mx = v; am = c; }       // first-max, like jnp.argmax
    }
    float se = 0.f;
    for (int c = 0; c < 24; ++c) se += __expf(outm[t * 24 + c] - mx);
    float lse = mx + __logf(se);
    int y = query_y[t];
    ce[t] = lse - outm[t * 24 + y];
    corr[t] = (am == y) ? 1 : 0;
  }
  __syncthreads();

  if (t == 0) {
    float cos_loss = 0.f;
    for (int gg = 0; gg < 120; ++gg) { float d = scos_g[gg] - 1.0f; cos_loss += d * d; }
    float sum_ce = 0.f; int acc = 0;
    for (int i = 0; i < 24; ++i) { sum_ce += ce[i]; acc += corr[i]; }
    float atten = ce[3];                    // maska picks row index ATT_CLASS=3
    float rest  = (sum_ce - atten) / 23.0f;
    out5[0] = atten + rest + cos_loss;
    out5[1] = (float)acc;
    out5[2] = cos_loss;
    out5[3] = atten;
    out5[4] = rest;
  }
}

// ---------------------------------------------------------------------------
extern "C" void kernel_launch(void* const* d_in, const int* in_sizes, int n_in,
                              void* d_out, int out_size, void* d_ws, size_t ws_size,
                              hipStream_t stream) {
  (void)n_in; (void)out_size; (void)ws_size;

  const float* support_x = (const float*)d_in[0];
  const int*   s_ei      = (const int*)d_in[1];
  const int*   s_batch   = (const int*)d_in[2];
  const int*   s_y       = (const int*)d_in[3];
  const float* query_x   = (const float*)d_in[4];
  const int*   q_batch   = (const int*)d_in[6];
  const int*   q_y       = (const int*)d_in[7];

  // params: detect flattening order (insertion: w0,g0,b0,...  vs sorted: b*,g*,w*)
  const float *Wp[6], *Gp[6], *Bp[6];
  if (in_sizes[8] > 100) {
    for (int l = 0; l < 6; ++l) {
      Wp[l] = (const float*)d_in[8 + 3 * l];
      Gp[l] = (const float*)d_in[9 + 3 * l];
      Bp[l] = (const float*)d_in[10 + 3 * l];
    }
  } else {
    for (int l = 0; l < 6; ++l) {
      Bp[l] = (const float*)d_in[8 + l];
      Gp[l] = (const float*)d_in[14 + l];
      Wp[l] = (const float*)d_in[20 + l];
    }
  }

  const int NS = in_sizes[0] / (3 * 64 * 64);
  const int NQ = in_sizes[4] / (3 * 64 * 64);
  const int ES = in_sizes[1] / 2;
  const int GS = 120, GQ = 24;

  // ---- workspace layout (floats; all region sizes are multiples of 8) ----
  float* W = (float*)d_ws;
  const long BUF0 = 251658240L;   // 1920*32*64*64 (max conv output)
  const long BUF1 = 62914560L;    // 1920*32*32*32 (max pooled output)
  float* buf0   = W;
  float* buf1   = buf0 + BUF0;
  float* sfeat  = buf1 + BUF1;                 // [NS,64]
  float* qfeat  = sfeat + (long)NS * 64;       // [NQ,64]
  float* ss     = qfeat + (long)NQ * 64;       // fused BN scale/shift (128)
  float* nrm    = ss + 128;                    // node norms [NS]
  float* wpkf   = nrm + NS;                    // repacked bf16 weights (<=73.7KB)
  float* zr     = wpkf + 18432;                // --- zeroed accumulators ---
  float* cossum = zr;
  float* coscnt = cossum + NS;
  float* s_cat  = coscnt + NS;                 // [GS,64]
  float* scnt   = s_cat + (long)GS * 64;
  float* scos   = scnt + GS;
  float* p_cat  = scos + GS;                   // [GQ,64]
  float* qcnt   = p_cat + (long)GQ * 64;
  const long ZRN = (qcnt + GQ) - zr;
  __bf16* wpk = (__bf16*)wpkf;

  zero_f32<<<(unsigned)((ZRN + 255) / 256), 256, 0, stream>>>(zr, ZRN);

  // ---- encoder: support then query (separate BN batch stats) ----
  const int chans[6][2] = {{3,32},{32,64},{64,64},{64,64},{64,64},{64,64}};
  for (int pass = 0; pass < 2; ++pass) {
    const int N = pass ? NQ : NS;
    const float* cur = pass ? query_x : support_x;
    int Hi = 64;
    for (int l = 0; l < 6; ++l) {
      const int Ci = chans[l][0], Co = chans[l][1];
      const int K  = Ci * 9;
      const int Kc = (K + 31) >> 5;
      const long wtot = (long)(Co >> 4) * Kc * 512;
      repack_w<<<(unsigned)((wtot + 255) / 256), 256, 0, stream>>>(Wp[l], wpk, Co, K);
      const long P = (long)N * Hi * Hi;
      conv3x3_wmma<<<(unsigned)((P + 63) / 64), 32 * (Co / 16), 0, stream>>>(
          cur, wpk, buf0, N, Ci, Co, Hi, Hi);
      bn_stats<<<Co, 256, 0, stream>>>(buf0, ss, Gp[l], Bp[l], N, Co, Hi * Hi);
      const long tot = (long)N * Co * (Hi / 2) * (Hi / 2);
      bn_relu_pool<<<(unsigned)((tot + 255) / 256), 256, 0, stream>>>(
          buf0, ss, buf1, N, Co, Hi, Hi);
      cur = buf1; Hi >>= 1;
    }
    const long fn = (long)N * 64;               // 1x1 spatial -> [N,64]
    copy_f32<<<(unsigned)((fn + 255) / 256), 256, 0, stream>>>(
        buf1, pass ? qfeat : sfeat, fn);
  }

  // ---- support edge cosines (query edges are dead code in the reference) ----
  node_norm<<<(NS + 255) / 256, 256, 0, stream>>>(sfeat, nrm, NS);
  edge_cos<<<(ES + 255) / 256, 256, 0, stream>>>(sfeat, nrm, s_ei, ES, cossum, coscnt);

  // ---- graph means ----
  support_graph_accum<<<(NS + 255) / 256, 256, 0, stream>>>(
      sfeat, cossum, coscnt, s_batch, NS, s_cat, scnt, scos);
  query_graph_accum<<<(NQ + 255) / 256, 256, 0, stream>>>(qfeat, q_batch, NQ, p_cat, qcnt);
  graph_div<<<(GS * 64 + 255) / 256, 256, 0, stream>>>(s_cat, scnt, scos, GS);
  graph_div<<<(GQ * 64 + 255) / 256, 256, 0, stream>>>(p_cat, qcnt, nullptr, GQ);

  // ---- losses / accuracy ----
  finalize<<<1, 256, 0, stream>>>(s_cat, p_cat, scos, s_y, q_y, (float*)d_out);
}
